// W8A16LinearLayer_26319559589917
// MI455X (gfx1250) — matrix operand order repaired and checked by reading, not compiled, not run on hardware
//
#include <hip/hip_runtime.h>
#include <stdint.h>

// ---- vector types matching gfx1250 WMMA builtin signatures ----
typedef _Float16 v16h __attribute__((ext_vector_type(16)));
typedef _Float16 v8h  __attribute__((ext_vector_type(8)));
typedef float    v8f  __attribute__((ext_vector_type(8)));
typedef int8_t   c8   __attribute__((ext_vector_type(8)));

#define IN_F  4096
#define OUT_F 11008
#define MTOT  4096   // B*S = 2*2048

constexpr int BM  = 128;
constexpr int BN  = 128;
constexpr int BK  = 32;
constexpr int PAD = 8;          // halfwords of padding per LDS row
constexpr int LDR = BK + PAD;   // 40 halfwords = 80 bytes row stride

union V16 { v16h v; v8h h[2]; };

__global__ __launch_bounds__(128)
void w8a16_gemm_wmma_f16(const float*  __restrict__ X,   // [MTOT, IN_F] fp32
                         const int8_t* __restrict__ W,   // [OUT_F, IN_F] int8
                         const float*  __restrict__ Sc,  // [OUT_F]
                         const float*  __restrict__ Bi,  // [OUT_F]
                         float*        __restrict__ Y)   // [MTOT, OUT_F] fp32
{
    // double-buffered tiles: 2 * (128*40 + 128*40) * 2B = 40 KB LDS
    __shared__ _Float16 sA[2][BM][LDR];   // activations, [m][k] f16
    __shared__ _Float16 sB[2][BN][LDR];   // weights,     [n][k] f16

    const int tid  = threadIdx.x;
    const int lane = tid & 31;
    const int wave = tid >> 5;          // 4 waves, 2x2 layout
    const int wm   = (wave >> 1) * 64;  // wave M offset in tile
    const int wn   = (wave &  1) * 64;  // wave N offset in tile
    const int row  = lane & 15;         // M-row (A) / N-col (B) within 16
    const int half = lane >> 4;         // K-half selector per ISA layout

    const int mBlock = blockIdx.y * BM;
    const int nBlock = blockIdx.x * BN;

    // staging decomposition: 4 threads cover one 32-wide K row
    const int tQ = tid & 3;   // k-octet: k = tQ*8 .. tQ*8+7
    const int tR = tid >> 2;  // row 0..31, 4 passes cover 128 rows

    // per-thread global base pointers for staging (advance by BK per step)
    const float*  xp = X + (size_t)(mBlock + tR) * IN_F + tQ * 8;
    const int8_t* wp = W + (size_t)(nBlock + tR) * IN_F + tQ * 8;

    v8f acc[4][4];
    {
        v8f z = {0.f, 0.f, 0.f, 0.f, 0.f, 0.f, 0.f, 0.f};
        #pragma unroll
        for (int i = 0; i < 4; ++i)
            #pragma unroll
            for (int j = 0; j < 4; ++j)
                acc[i][j] = z;
    }

    // ---- stage one (BM x BK) A-tile + (BN x BK) B-tile into buffer `b` ----
    auto stage = [&](int k0, int b) {
        #pragma unroll
        for (int p = 0; p < 4; ++p) {
            const int m = tR + p * 32;
            const float4* src =
                (const float4*)(xp + (size_t)p * 32 * IN_F + k0);
            float4 a0 = src[0];
            float4 a1 = src[1];
            v8h h;
            h[0] = (_Float16)a0.x; h[1] = (_Float16)a0.y;
            h[2] = (_Float16)a0.z; h[3] = (_Float16)a0.w;
            h[4] = (_Float16)a1.x; h[5] = (_Float16)a1.y;
            h[6] = (_Float16)a1.z; h[7] = (_Float16)a1.w;
            *(v8h*)&sA[b][m][tQ * 8] = h;   // 16B aligned: 80*m + 16*tQ
        }
        #pragma unroll
        for (int p = 0; p < 4; ++p) {
            const int n = tR + p * 32;
            c8 w = *(const c8*)(wp + (size_t)p * 32 * IN_F + k0);
            v8h h;
            #pragma unroll
            for (int e = 0; e < 8; ++e) h[e] = (_Float16)w[e];  // int8 exact in f16
            *(v8h*)&sB[b][n][tQ * 8] = h;
        }
    };

    // ---- 16 WMMAs over buffer `b` ----
    auto compute = [&](int b) {
        V16 bf[4];
        #pragma unroll
        for (int j = 0; j < 4; ++j) {
            const _Float16* pb = &sB[b][wn + j * 16 + row][half * 8];
            bf[j].h[0] = *(const v8h*)(pb);      // K = half*8 .. +7
            bf[j].h[1] = *(const v8h*)(pb + 16); // K = 16+half*8 .. +7
        }
        #pragma unroll
        for (int i = 0; i < 4; ++i) {
            V16 af;
            const _Float16* pa = &sA[b][wm + i * 16 + row][half * 8];
            af.h[0] = *(const v8h*)(pa);        // K = half*8 .. +7
            af.h[1] = *(const v8h*)(pa + 16);   // K = 16+half*8 .. +7
            #pragma unroll
            for (int j = 0; j < 4; ++j) {
                acc[i][j] = __builtin_amdgcn_wmma_f32_16x16x32_f16(
                    /*neg_a=*/false, af.v,
                    /*neg_b=*/false, bf[j].v,
                    /*c_mod=*/(short)0, acc[i][j],
                    /*reuse_a=*/false, /*reuse_b=*/false);
            }
        }
    };

    // ---- double-buffered main loop: one barrier per K-step ----
    stage(0, 0);
    int buf = 0;
    #pragma unroll 1
    for (int k0 = 0; k0 < IN_F - BK; k0 += BK) {
        __syncthreads();
        stage(k0 + BK, buf ^ 1);   // fill other buffer while computing this one
        compute(buf);
        buf ^= 1;
    }
    __syncthreads();
    compute(buf);

    // ---- epilogue: per-output-channel scale + bias ----
    // Y is written once and never re-read; use non-temporal stores so the
    // 180 MB of output traffic streams through without evicting the X (64 MB)
    // + W (45 MB) working set from the 192 MB L2 (inter-block reuse: each X
    // tile is read by 86 blocks, each W strip by 32 blocks).
    float scj[4], bij[4];
    #pragma unroll
    for (int j = 0; j < 4; ++j) {
        const int gn = nBlock + wn + j * 16 + row;
        scj[j] = Sc[gn];
        bij[j] = Bi[gn];
    }
    #pragma unroll
    for (int i = 0; i < 4; ++i) {
        // C/D layout: lane = N col; VGPR r = M row (r + 8*half)
        const int gm0 = mBlock + wm + i * 16 + half * 8;
        #pragma unroll
        for (int j = 0; j < 4; ++j) {
            float* yp = Y + (size_t)gm0 * OUT_F + (nBlock + wn + j * 16 + row);
            #pragma unroll
            for (int r = 0; r < 8; ++r) {
                __builtin_nontemporal_store(acc[i][j][r] * scj[j] + bij[j],
                                            yp + (size_t)r * OUT_F);
            }
        }
    }
}

extern "C" void kernel_launch(void* const* d_in, const int* in_sizes, int n_in,
                              void* d_out, int out_size, void* d_ws, size_t ws_size,
                              hipStream_t stream)
{
    const float*  X  = (const float*)d_in[0];
    const int8_t* W  = (const int8_t*)d_in[1];
    const float*  Sc = (const float*)d_in[2];
    const float*  Bi = (const float*)d_in[3];
    float*        Y  = (float*)d_out;

    dim3 grid(OUT_F / BN, MTOT / BM, 1);  // (86, 32)
    w8a16_gemm_wmma_f16<<<grid, 128, 0, stream>>>(X, W, Sc, Bi, Y);
}
